// GNNEncoder_36034775614223
// MI455X (gfx1250) — matrix operand order repaired and checked by reading, not compiled, not run on hardware
//
#include <hip/hip_runtime.h>

// ---------------- constants ----------------
constexpr int NN   = 8192;            // nodes
constexpr int NE   = 131072;          // edges
constexpr int ENR  = NE + NN;         // edges + self loops = 139264
constexpr int NB   = 64;              // batches
constexpr int DD   = 128;             // feature dim
constexpr int NH   = 4;               // heads
constexpr int HD   = NH * DD;         // 512
constexpr int NOUT = 256;

typedef _Float16 half_t;
typedef _Float16 v16h __attribute__((ext_vector_type(16)));
typedef _Float16 v8h  __attribute__((ext_vector_type(8)));
typedef float    v8f  __attribute__((ext_vector_type(8)));

// ---------------- helpers ----------------
__device__ __forceinline__ float wave_sum(float v) {
    #pragma unroll
    for (int off = 16; off > 0; off >>= 1) v += __shfl_xor(v, off, 32);
    return v;
}
// monotonic float<->uint key for atomicMax on floats
__device__ __forceinline__ unsigned f2key(float f) {
    unsigned b = __float_as_uint(f);
    return (b & 0x80000000u) ? ~b : (b | 0x80000000u);
}
__device__ __forceinline__ float key2f(unsigned k) {
    unsigned b = (k & 0x80000000u) ? (k & 0x7FFFFFFFu) : ~k;
    return __uint_as_float(b);
}
__device__ __forceinline__ float sigf(float x) { return 1.0f / (1.0f + expf(-x)); }

// ---------------- fills ----------------
__global__ void k_fill_f(float* p, float v, int n) {
    int i = blockIdx.x * blockDim.x + threadIdx.x;
    if (i < n) p[i] = v;
}
__global__ void k_fill_u(unsigned* p, unsigned v, int n) {
    int i = blockIdx.x * blockDim.x + threadIdx.x;
    if (i < n) p[i] = v;
}

// ---------------- embeddings ----------------
// h[n,d] = x[n,:8] @ node_w[:, d] + node_b[d]
__global__ void k_node_embed(const float* __restrict__ x, const float* __restrict__ w,
                             const float* __restrict__ b, float* __restrict__ h) {
    int idx = blockIdx.x * blockDim.x + threadIdx.x;
    if (idx >= NN * DD) return;
    int n = idx >> 7, d = idx & 127;
    float acc = b[d];
    #pragma unroll
    for (int k = 0; k < 8; ++k) acc += x[n * 8 + k] * w[k * DD + d];
    h[idx] = acc;
}

// ea[e,:] = l2norm(edge_attr[e,:2] @ edge_w + edge_b), also accumulate column sums
__global__ void k_edge_embed(const float* __restrict__ eattr, const float* __restrict__ ew,
                             const float* __restrict__ eb, float* __restrict__ ea,
                             float* __restrict__ eam) {
    __shared__ float sm[DD];
    int lane = threadIdx.x & 31, wid = threadIdx.x >> 5;
    if (threadIdx.x < DD) sm[threadIdx.x] = 0.0f;
    __syncthreads();
    int e = blockIdx.x * 8 + wid;
    float a0 = eattr[e * 2 + 0], a1 = eattr[e * 2 + 1];
    int d0 = lane * 4;
    float v[4]; float ss = 0.f;
    #pragma unroll
    for (int j = 0; j < 4; ++j) {
        float t = a0 * ew[d0 + j] + a1 * ew[DD + d0 + j] + eb[d0 + j];
        v[j] = t; ss += t * t;
    }
    ss = wave_sum(ss);
    float inv = 1.0f / fmaxf(sqrtf(ss), 1e-12f);
    #pragma unroll
    for (int j = 0; j < 4; ++j) {
        float t = v[j] * inv;
        ea[(size_t)e * DD + d0 + j] = t;
        atomicAdd(&sm[d0 + j], t);
    }
    __syncthreads();
    if (threadIdx.x < DD) atomicAdd(&eam[threadIdx.x], sm[threadIdx.x]);
}

// self-loop rows: ea[E+n, d] = mean_e ea[e, d]
__global__ void k_selfloop(const float* __restrict__ eam, float* __restrict__ ea) {
    int idx = blockIdx.x * blockDim.x + threadIdx.x;
    if (idx >= NN * DD) return;
    ea[(size_t)NE * DD + idx] = eam[idx & 127] * (1.0f / NE);
}

// ---------------- per-layer prep ----------------
// wf[c,h] = sum_d lin_edge_w[c, h*128+d] * att_edge[h, d]
__global__ void k_wfold(const float* __restrict__ lew, const float* __restrict__ ae,
                        float* __restrict__ wf) {
    int idx = blockIdx.x * blockDim.x + threadIdx.x;
    if (idx >= DD * NH) return;
    int c = idx >> 2, hd = idx & 3;
    float acc = 0.f;
    for (int d = 0; d < DD; ++d) acc += lew[c * HD + hd * DD + d] * ae[hd * DD + d];
    wf[idx] = acc;
}

// transpose+convert weights: wt[n,k] = (f16) lin_w[k,n]
__global__ void k_conv_wt(const float* __restrict__ lw, half_t* __restrict__ wt) {
    int idx = blockIdx.x * blockDim.x + threadIdx.x;
    if (idx >= HD * DD) return;
    int n = idx >> 7, k = idx & 127;
    wt[idx] = (half_t)lw[k * HD + n];
}

__global__ void k_conv_h(const float* __restrict__ h, half_t* __restrict__ hf) {
    int idx = blockIdx.x * blockDim.x + threadIdx.x;
    if (idx < NN * DD) hf[idx] = (half_t)h[idx];
}

// ---------------- WMMA GEMM: xh[8192,512] = hf16[8192,128] @ Wt^T ----------------
// Bt is [512][128] f16 (row = output column, contiguous K). One 16x16 tile per wave.
__global__ __launch_bounds__(256) void k_gemm_xh(const half_t* __restrict__ A,
                                                 const half_t* __restrict__ Bt,
                                                 float* __restrict__ Xh) {
    const int lane = threadIdx.x & 31;
    const int wid  = threadIdx.x >> 5;
    const int bn = blockIdx.x & 31;          // 32 column tiles (512/16)
    const int bm = blockIdx.x >> 5;          // 64 row groups of 8 tiles
    const int m0 = (bm * 8 + wid) * 16;
    const int n0 = bn * 16;
    const int r  = lane & 15;
    const int abase = (lane >> 4) * 8;       // A: interleaved K layout
    const int bbase = (lane >> 4) * 8;       // B: interleaved K layout
    const half_t* ap = A  + (size_t)(m0 + r) * DD + abase;
    const half_t* bp = Bt + (size_t)(n0 + r) * DD + bbase;
    v8f c = {};
    #pragma unroll
    for (int kk = 0; kk < DD; kk += 32) {
        v8h a0 = *(const v8h*)(ap + kk);
        v8h a1 = *(const v8h*)(ap + kk + 16);
        v8h b0 = *(const v8h*)(bp + kk);
        v8h b1 = *(const v8h*)(bp + kk + 16);
        v16h av = __builtin_shufflevector(a0, a1, 0,1,2,3,4,5,6,7,8,9,10,11,12,13,14,15);
        v16h bv = __builtin_shufflevector(b0, b1, 0,1,2,3,4,5,6,7,8,9,10,11,12,13,14,15);
        c = __builtin_amdgcn_wmma_f32_16x16x32_f16(false, av, false, bv,
                                                   (short)0, c, false, false);
    }
    float* outp = Xh + (size_t)(m0 + abase) * HD + (n0 + r);
    #pragma unroll
    for (int i = 0; i < 8; ++i) outp[i * HD] = c[i];
}

// ---------------- attention logits ----------------
// a_s[n,h] = xh[n,h,:]·att_src[h], a_d likewise. One wave per node.
__global__ void k_attn_sd(const float* __restrict__ xh, const float* __restrict__ asrc,
                          const float* __restrict__ adst, float* __restrict__ a_s,
                          float* __restrict__ a_d) {
    int lane = threadIdx.x & 31, wid = threadIdx.x >> 5;
    int n = blockIdx.x * 8 + wid;
    int d0 = lane * 4;
    #pragma unroll
    for (int hd = 0; hd < NH; ++hd) {
        float ps = 0.f, pd = 0.f;
        #pragma unroll
        for (int j = 0; j < 4; ++j) {
            float xv = xh[(size_t)n * HD + hd * DD + d0 + j];
            ps += xv * asrc[hd * DD + d0 + j];
            pd += xv * adst[hd * DD + d0 + j];
        }
        ps = wave_sum(ps); pd = wave_sum(pd);
        if (lane == 0) { a_s[n * NH + hd] = ps; a_d[n * NH + hd] = pd; }
    }
}

// a_e[i,h] = ea[i,:]·wf[:,h]   (folded edge projection). One wave per row.
__global__ void k_attn_e(const float* __restrict__ ea, const float* __restrict__ wf,
                         float* __restrict__ a_e) {
    int lane = threadIdx.x & 31, wid = threadIdx.x >> 5;
    int i = blockIdx.x * 8 + wid;
    int d0 = lane * 4;
    float acc0 = 0, acc1 = 0, acc2 = 0, acc3 = 0;
    #pragma unroll
    for (int j = 0; j < 4; ++j) {
        float ev = ea[(size_t)i * DD + d0 + j];
        const float* w = wf + (d0 + j) * NH;
        acc0 += ev * w[0]; acc1 += ev * w[1]; acc2 += ev * w[2]; acc3 += ev * w[3];
    }
    acc0 = wave_sum(acc0); acc1 = wave_sum(acc1);
    acc2 = wave_sum(acc2); acc3 = wave_sum(acc3);
    if (lane == 0) {
        a_e[i * NH + 0] = acc0; a_e[i * NH + 1] = acc1;
        a_e[i * NH + 2] = acc2; a_e[i * NH + 3] = acc3;
    }
}

// ---------------- segment softmax over dst ----------------
__global__ void k_alpha_max(const int* __restrict__ eidx, const float* __restrict__ a_s,
                            const float* __restrict__ a_d, const float* __restrict__ a_e,
                            float* __restrict__ alpha, unsigned* __restrict__ smax) {
    int idx = blockIdx.x * blockDim.x + threadIdx.x;
    if (idx >= ENR * NH) return;
    int i = idx >> 2, hd = idx & 3;
    int s, t;
    if (i < NE) { s = eidx[i]; t = eidx[NE + i]; } else { s = t = i - NE; }
    float a = a_s[s * NH + hd] + a_d[t * NH + hd] + a_e[idx];
    a = (a > 0.f) ? a : 0.2f * a;                 // leaky_relu(0.2)
    alpha[idx] = a;
    atomicMax(&smax[t * NH + hd], f2key(a));
}

__global__ void k_exp_sum(const int* __restrict__ eidx, float* __restrict__ alpha,
                          const unsigned* __restrict__ smax, float* __restrict__ sden) {
    int idx = blockIdx.x * blockDim.x + threadIdx.x;
    if (idx >= ENR * NH) return;
    int i = idx >> 2, hd = idx & 3;
    int t = (i < NE) ? eidx[NE + i] : i - NE;
    float ex = expf(alpha[idx] - key2f(smax[t * NH + hd]));
    alpha[idx] = ex;
    atomicAdd(&sden[t * NH + hd], ex);
}

// out[dst,h,:] += xh[src,h,:] * att.  One wave per edge row.
__global__ void k_scatter(const int* __restrict__ eidx, const float* __restrict__ xh,
                          const float* __restrict__ alpha, const float* __restrict__ sden,
                          float* __restrict__ ob) {
    int lane = threadIdx.x & 31, wid = threadIdx.x >> 5;
    int i = blockIdx.x * 8 + wid;
    int s, t;
    if (i < NE) { s = eidx[i]; t = eidx[NE + i]; } else { s = t = i - NE; }
    int d0 = lane * 4;
    #pragma unroll
    for (int hd = 0; hd < NH; ++hd) {
        float att = alpha[i * NH + hd] / (sden[t * NH + hd] + 1e-16f);
        #pragma unroll
        for (int j = 0; j < 4; ++j) {
            int off = hd * DD + d0 + j;
            atomicAdd(&ob[(size_t)t * HD + off], xh[(size_t)s * HD + off] * att);
        }
    }
}

// head-mean + bias, accumulate batchnorm stats. blockDim=128 (d), each block 64 nodes.
__global__ void k_headmean(const float* __restrict__ ob, const float* __restrict__ bias,
                           float* __restrict__ g, float* __restrict__ bnsum,
                           float* __restrict__ bnsq) {
    int d = threadIdx.x;
    float s = 0.f, q = 0.f;
    for (int i = 0; i < 64; ++i) {
        int n = blockIdx.x * 64 + i;
        const float* row = ob + (size_t)n * HD;
        float v = 0.25f * (row[d] + row[DD + d] + row[2 * DD + d] + row[3 * DD + d]) + bias[d];
        g[(size_t)n * DD + d] = v;
        s += v; q += v * v;
    }
    atomicAdd(&bnsum[d], s);
    atomicAdd(&bnsq[d], q);
}

// batchnorm + relu + residual (h updated in place)
__global__ void k_bn_apply(const float* __restrict__ g, const float* __restrict__ bnsum,
                           const float* __restrict__ bnsq, const float* __restrict__ gamma,
                           const float* __restrict__ beta, float* __restrict__ h) {
    int idx = blockIdx.x * blockDim.x + threadIdx.x;
    if (idx >= NN * DD) return;
    int d = idx & 127;
    float mu  = bnsum[d] * (1.0f / NN);
    float var = bnsq[d] * (1.0f / NN) - mu * mu;
    float y = gamma[d] * (g[idx] - mu) * rsqrtf(var + 1e-5f) + beta[d];
    h[idx] = fmaxf(y, 0.f) + h[idx];
}

// ---------------- Set2Set ----------------
__global__ void k_gates(const float* __restrict__ qs, const float* __restrict__ hs,
                        const float* __restrict__ wih, const float* __restrict__ whh,
                        const float* __restrict__ bih, const float* __restrict__ bhh,
                        float* __restrict__ gates) {
    int idx = blockIdx.x * blockDim.x + threadIdx.x;
    if (idx >= NB * 4 * DD) return;
    int b = idx >> 9, j = idx & 511;
    float acc = bih[j] + bhh[j];
    const float* q = qs + b * 2 * DD;
    const float* wi = wih + j * 2 * DD;
    for (int k = 0; k < 2 * DD; ++k) acc += q[k] * wi[k];
    const float* hh = hs + b * DD;
    const float* wh = whh + j * DD;
    for (int k = 0; k < DD; ++k) acc += hh[k] * wh[k];
    gates[idx] = acc;
}

__global__ void k_lstm_up(const float* __restrict__ gates, float* __restrict__ cs,
                          float* __restrict__ hs) {
    int idx = blockIdx.x * blockDim.x + threadIdx.x;
    if (idx >= NB * DD) return;
    int b = idx >> 7, d = idx & 127;
    const float* gr = gates + b * 4 * DD;
    float gi = sigf(gr[d]), gf = sigf(gr[DD + d]);
    float gg = tanhf(gr[2 * DD + d]), go = sigf(gr[3 * DD + d]);
    float c = gf * cs[idx] + gi * gg;
    cs[idx] = c;
    hs[idx] = go * tanhf(c);
}

__global__ void k_s2s_e(const float* __restrict__ h, const float* __restrict__ hs,
                        const int* __restrict__ batch, float* __restrict__ eN,
                        unsigned* __restrict__ smax) {
    int lane = threadIdx.x & 31, wid = threadIdx.x >> 5;
    int n = blockIdx.x * 8 + wid;
    int b = batch[n];
    int d0 = lane * 4;
    float p = 0.f;
    #pragma unroll
    for (int j = 0; j < 4; ++j) p += h[(size_t)n * DD + d0 + j] * hs[b * DD + d0 + j];
    p = wave_sum(p);
    if (lane == 0) { eN[n] = p; atomicMax(&smax[b], f2key(p)); }
}

__global__ void k_s2s_exp(const int* __restrict__ batch, const float* __restrict__ eN,
                          const unsigned* __restrict__ smax, float* __restrict__ aN,
                          float* __restrict__ sden) {
    int n = blockIdx.x * blockDim.x + threadIdx.x;
    if (n >= NN) return;
    int b = batch[n];
    float ex = expf(eN[n] - key2f(smax[b]));
    aN[n] = ex;
    atomicAdd(&sden[b], ex);
}

__global__ void k_s2s_r(const int* __restrict__ batch, const float* __restrict__ h,
                        const float* __restrict__ aN, const float* __restrict__ sden,
                        float* __restrict__ rB) {
    int idx = blockIdx.x * blockDim.x + threadIdx.x;
    if (idx >= NN * DD) return;
    int n = idx >> 7, d = idx & 127;
    int b = batch[n];
    atomicAdd(&rB[b * DD + d], (aN[n] / (sden[b] + 1e-16f)) * h[idx]);
}

__global__ void k_qstar(const float* __restrict__ hs, const float* __restrict__ rB,
                        float* __restrict__ qs) {
    int idx = blockIdx.x * blockDim.x + threadIdx.x;
    if (idx >= NB * DD) return;
    int b = idx >> 7, d = idx & 127;
    qs[b * 2 * DD + d] = hs[idx];
    qs[b * 2 * DD + DD + d] = rB[idx];
}

// ---------------- output MLP ----------------
__global__ void k_mlp1(const float* __restrict__ qs, const float* __restrict__ w1,
                       const float* __restrict__ b1, float* __restrict__ z1) {
    int idx = blockIdx.x * blockDim.x + threadIdx.x;
    if (idx >= NB * DD) return;
    int b = idx >> 7, j = idx & 127;
    float acc = b1[j];
    for (int k = 0; k < 2 * DD; ++k) acc += qs[b * 2 * DD + k] * w1[k * DD + j];
    z1[idx] = fmaxf(acc, 0.f);
}

__global__ void k_mlp2(const float* __restrict__ z1, const float* __restrict__ w2,
                       const float* __restrict__ b2, float* __restrict__ z2) {
    int idx = blockIdx.x * blockDim.x + threadIdx.x;
    if (idx >= NB * NOUT) return;
    int b = idx >> 8, j = idx & 255;
    float acc = b2[j];
    for (int k = 0; k < DD; ++k) acc += z1[b * DD + k] * w2[k * NOUT + j];
    z2[idx] = fmaxf(acc, 0.f);
}

__global__ void k_outnorm(const float* __restrict__ z2, float* __restrict__ outp) {
    __shared__ float sred[8];
    __shared__ float snorm;
    int b = blockIdx.x, t = threadIdx.x;
    int lane = t & 31, wid = t >> 5;
    float v = z2[b * NOUT + t];
    float ss = wave_sum(v * v);
    if (lane == 0) sred[wid] = ss;
    __syncthreads();
    if (t == 0) {
        float tot = 0.f;
        for (int i = 0; i < 8; ++i) tot += sred[i];
        snorm = fmaxf(sqrtf(tot), 1e-12f);
    }
    __syncthreads();
    outp[b * NOUT + t] = v / snorm;
}

// ---------------- host ----------------
extern "C" void kernel_launch(void* const* d_in, const int* in_sizes, int n_in,
                              void* d_out, int out_size, void* d_ws, size_t ws_size,
                              hipStream_t stream) {
    const float* x      = (const float*)d_in[0];
    const int*   eidx   = (const int*)d_in[1];
    const float* eattr  = (const float*)d_in[2];
    const int*   batch  = (const int*)d_in[3];
    const float* node_w = (const float*)d_in[4];
    const float* node_b = (const float*)d_in[5];
    const float* edge_w = (const float*)d_in[6];
    const float* edge_b = (const float*)d_in[7];
    const float* lin_w  = (const float*)d_in[8];
    const float* att_s  = (const float*)d_in[9];
    const float* att_d  = (const float*)d_in[10];
    const float* lin_ew = (const float*)d_in[11];
    const float* att_e  = (const float*)d_in[12];
    const float* gbias  = (const float*)d_in[13];
    const float* bng    = (const float*)d_in[14];
    const float* bnb    = (const float*)d_in[15];
    const float* wih    = (const float*)d_in[16];
    const float* whh    = (const float*)d_in[17];
    const float* bih    = (const float*)d_in[18];
    const float* bhh    = (const float*)d_in[19];
    const float* w1     = (const float*)d_in[20];
    const float* b1     = (const float*)d_in[21];
    const float* w2     = (const float*)d_in[22];
    const float* b2     = (const float*)d_in[23];
    float* outp = (float*)d_out;

    char* wsb = (char*)d_ws;
    size_t cur = 0;
    auto alloc = [&](size_t bytes) -> char* {
        char* p = wsb + cur;
        cur = (cur + bytes + 255) & ~((size_t)255);
        return p;
    };
    float*    h     = (float*)alloc((size_t)NN * DD * 4);
    float*    ea    = (float*)alloc((size_t)ENR * DD * 4);
    float*    xh    = (float*)alloc((size_t)NN * HD * 4);
    float*    ob    = (float*)alloc((size_t)NN * HD * 4);
    float*    g     = (float*)alloc((size_t)NN * DD * 4);
    float*    a_s   = (float*)alloc((size_t)NN * NH * 4);
    float*    a_d   = (float*)alloc((size_t)NN * NH * 4);
    float*    a_e   = (float*)alloc((size_t)ENR * NH * 4);
    float*    alpha = (float*)alloc((size_t)ENR * NH * 4);
    unsigned* smaxE = (unsigned*)alloc((size_t)NN * NH * 4);
    float*    sdenE = (float*)alloc((size_t)NN * NH * 4);
    float*    bnsum = (float*)alloc(DD * 4);
    float*    bnsq  = (float*)alloc(DD * 4);
    float*    eam   = (float*)alloc(DD * 4);
    float*    wfold = (float*)alloc(DD * NH * 4);
    half_t*   wt    = (half_t*)alloc((size_t)HD * DD * 2);
    half_t*   hf    = (half_t*)alloc((size_t)NN * DD * 2);
    float*    hs    = (float*)alloc(NB * DD * 4);
    float*    cs    = (float*)alloc(NB * DD * 4);
    float*    qs    = (float*)alloc(NB * 2 * DD * 4);
    float*    gates = (float*)alloc(NB * 4 * DD * 4);
    float*    eN    = (float*)alloc(NN * 4);
    float*    aN    = (float*)alloc(NN * 4);
    float*    rB    = (float*)alloc(NB * DD * 4);
    unsigned* smaxB = (unsigned*)alloc(NB * 4);
    float*    sdenB = (float*)alloc(NB * 4);
    float*    z1    = (float*)alloc(NB * DD * 4);
    float*    z2    = (float*)alloc(NB * NOUT * 4);

    dim3 blk(256);
    #define GRID(n) dim3((unsigned)(((n) + 255) / 256))

    // embeddings
    k_fill_f<<<1, blk, 0, stream>>>(eam, 0.f, DD);
    k_node_embed<<<GRID(NN * DD), blk, 0, stream>>>(x, node_w, node_b, h);
    k_edge_embed<<<NE / 8, blk, 0, stream>>>(eattr, edge_w, edge_b, ea, eam);
    k_selfloop<<<GRID(NN * DD), blk, 0, stream>>>(eam, ea);

    // 3 GAT layers
    for (int l = 0; l < 3; ++l) {
        const float* lw  = lin_w  + (size_t)l * DD * HD;
        const float* lew = lin_ew + (size_t)l * DD * HD;
        const float* asW = att_s + l * HD;
        const float* adW = att_d + l * HD;
        const float* aeW = att_e + l * HD;
        const float* biW = gbias + l * DD;
        const float* gW  = bng + l * DD;
        const float* bW  = bnb + l * DD;

        k_fill_f<<<GRID(NN * HD), blk, 0, stream>>>(ob, 0.f, NN * HD);
        k_fill_u<<<GRID(NN * NH), blk, 0, stream>>>(smaxE, 0u, NN * NH);
        k_fill_f<<<GRID(NN * NH), blk, 0, stream>>>(sdenE, 0.f, NN * NH);
        k_fill_f<<<1, blk, 0, stream>>>(bnsum, 0.f, DD);
        k_fill_f<<<1, blk, 0, stream>>>(bnsq, 0.f, DD);

        k_wfold<<<2, blk, 0, stream>>>(lew, aeW, wfold);
        k_conv_wt<<<GRID(HD * DD), blk, 0, stream>>>(lw, wt);
        k_conv_h<<<GRID(NN * DD), blk, 0, stream>>>(h, hf);
        k_gemm_xh<<<2048, blk, 0, stream>>>(hf, wt, xh);

        k_attn_sd<<<NN / 8, blk, 0, stream>>>(xh, asW, adW, a_s, a_d);
        k_attn_e<<<ENR / 8, blk, 0, stream>>>(ea, wfold, a_e);
        k_alpha_max<<<GRID(ENR * NH), blk, 0, stream>>>(eidx, a_s, a_d, a_e, alpha, smaxE);
        k_exp_sum<<<GRID(ENR * NH), blk, 0, stream>>>(eidx, alpha, smaxE, sdenE);
        k_scatter<<<ENR / 8, blk, 0, stream>>>(eidx, xh, alpha, sdenE, ob);
        k_headmean<<<128, dim3(128), 0, stream>>>(ob, biW, g, bnsum, bnsq);
        k_bn_apply<<<GRID(NN * DD), blk, 0, stream>>>(g, bnsum, bnsq, gW, bW, h);
    }

    // Set2Set, 3 processing steps
    k_fill_f<<<GRID(NB * DD), blk, 0, stream>>>(hs, 0.f, NB * DD);
    k_fill_f<<<GRID(NB * DD), blk, 0, stream>>>(cs, 0.f, NB * DD);
    k_fill_f<<<GRID(NB * 2 * DD), blk, 0, stream>>>(qs, 0.f, NB * 2 * DD);
    for (int s = 0; s < 3; ++s) {
        k_gates<<<GRID(NB * 4 * DD), blk, 0, stream>>>(qs, hs, wih, whh, bih, bhh, gates);
        k_lstm_up<<<GRID(NB * DD), blk, 0, stream>>>(gates, cs, hs);
        k_fill_u<<<1, blk, 0, stream>>>(smaxB, 0u, NB);
        k_fill_f<<<1, blk, 0, stream>>>(sdenB, 0.f, NB);
        k_fill_f<<<GRID(NB * DD), blk, 0, stream>>>(rB, 0.f, NB * DD);
        k_s2s_e<<<NN / 8, blk, 0, stream>>>(h, hs, batch, eN, smaxB);
        k_s2s_exp<<<GRID(NN), blk, 0, stream>>>(batch, eN, smaxB, aN, sdenB);
        k_s2s_r<<<GRID(NN * DD), blk, 0, stream>>>(batch, h, aN, sdenB, rB);
        k_qstar<<<GRID(NB * DD), blk, 0, stream>>>(hs, rB, qs);
    }

    // output MLP + row l2norm
    k_mlp1<<<GRID(NB * DD), blk, 0, stream>>>(qs, w1, b1, z1);
    k_mlp2<<<GRID(NB * NOUT), blk, 0, stream>>>(z1, w2, b2, z2);
    k_outnorm<<<NB, blk, 0, stream>>>(z2, outp);
    #undef GRID
}
